// IJEPADepthSmart_88304527606229
// MI455X (gfx1250) — matrix operand order repaired and checked by reading, not compiled, not run on hardware
//
#include <hip/hip_runtime.h>
#include <hip/hip_bf16.h>
#include <math.h>

// ---------------- problem constants ----------------
#define LNUM 10
#define DIM 768
#define HIDN 3072
#define NHEAD 12
#define DHEAD 64
#define BATCH 4
#define SEQ 1024
#define NTOK (BATCH * SEQ)   // 4096
#define MASKID (-100)

// gemm epilogue flags
#define FLAG_BIAS 1
#define FLAG_RES  2
#define FLAG_GELU 4
#define FLAG_OUTBF16 8

typedef __attribute__((ext_vector_type(16))) __bf16 v16bf;
typedef __attribute__((ext_vector_type(8)))  __bf16 v8bf;
typedef __attribute__((ext_vector_type(8)))  float  v8f;
typedef __attribute__((ext_vector_type(4)))  unsigned int u32x4;
typedef __attribute__((ext_vector_type(8)))  int i32x8;
typedef __attribute__((ext_vector_type(4)))  int i32x4;

#if __has_builtin(__builtin_amdgcn_tensor_load_to_lds) && __has_builtin(__builtin_amdgcn_s_wait_tensorcnt)
#define USE_TDM 1
#else
#define USE_TDM 0
#endif

// fp32 -> bf16 via hardware convert (v_cvt_*_bf16_f32 class, RNE)
__device__ __forceinline__ __bf16 f2bf(float f) {
    return (__bf16)f;
}

__device__ __forceinline__ float gelu_exact(float v) {
    return 0.5f * v * (1.0f + erff(v * 0.70710678118654752f));
}

__device__ __forceinline__ v16bf frag16(const __bf16* p0, const __bf16* p1) {
    v8bf lo = *(const v8bf*)p0;
    v8bf hi = *(const v8bf*)p1;
    return __builtin_shufflevector(lo, hi, 0, 1, 2, 3, 4, 5, 6, 7,
                                           8, 9, 10, 11, 12, 13, 14, 15);
}

// ---------------------------------------------------------------
// GEMM: C[M,N] = bf16A[M,K] @ f32W[K,N] (+bias)(+gelu)(+residual)
// bf16 WMMA 16x16x32, f32 accumulate. Block tile 128x64 (8 waves),
// each wave computes 32x32 (4 accumulators). A tile staged to LDS
// by the Tensor Data Mover (pad feature gives the 40-half stride);
// W tile converted fp32->bf16 and stored transposed [n][k] so each
// lane's B fragment is two contiguous 16B LDS loads.
// M % 128 == 0, N % 64 == 0, K % 32 == 0 in this workload.
// ---------------------------------------------------------------
__global__ __launch_bounds__(256) void gemm_bf16_wmma_kernel(
    const __bf16* __restrict__ A, const float* __restrict__ W,
    const float* __restrict__ bias, const float* __restrict__ resid,
    float* __restrict__ C, __bf16* __restrict__ Cb,
    int M, int N, int K, int flags)
{
    __shared__ __align__(16) __bf16 As[128][40];   // [m][k], stride 40 halves
    __shared__ __align__(16) __bf16 Bs[64][40];    // [n][k] transposed

    const int tid  = threadIdx.x;
    const int m0   = blockIdx.y * 128;
    const int n0   = blockIdx.x * 64;
    const int lane = tid & 31;
    const int wid  = tid >> 5;
    const int wy   = wid >> 1;      // 0..3 -> 32-row slab
    const int wx   = wid & 1;       // 0..1 -> 32-col slab
    const int hlf  = lane >> 4;
    const int ml   = lane & 15;

    v8f acc00 = {}, acc01 = {}, acc10 = {}, acc11 = {};

    // W staging: thread -> (k = tid>>3, n = (tid&7)*8 .. +7)
    const int kk = tid >> 3, nn = (tid & 7) * 8;
#if !USE_TDM
    const int r2 = tid >> 1, c2 = (tid & 1) * 16;  // A fallback copy
#endif

    for (int k0 = 0; k0 < K; k0 += 32) {
#if USE_TDM
        if (wid == 0) {
            // Tensor DMA: 128x32 bf16 tile, LDS pad 16B every 64B -> stride 40 halves
            unsigned long long ga = (unsigned long long)(size_t)(A + (size_t)m0 * K + k0);
            unsigned ldsa = (unsigned)(size_t)(&As[0][0]);
            u32x4 g0;
            g0[0] = 1u;                                   // count=1, user mode
            g0[1] = ldsa;                                 // lds_addr
            g0[2] = (unsigned)(ga & 0xFFFFFFFFu);         // global_addr[31:0]
            g0[3] = (unsigned)((ga >> 32) & 0x1FFFFFFu)   // global_addr[56:32]
                    | (2u << 30);                         // type=2 (image)
            i32x8 g1;
            g1[0] = (1 << 16)        // data_size = 2 bytes
                  | (1 << 20)        // pad_enable
                  | (3 << 22)        // pad_interval: 16 DWORDs (64B)
                  | (3 << 25);       // pad_amount: 4 DWORDs (16B)
            g1[1] = (32  << 16);     // tensor_dim0 = 32 (low16 @ bits 63:48)
            g1[2] = (128 << 16);     // dim0 hi=0 | tensor_dim1 = 128 (low16 @ 95:80)
            g1[3] = (32  << 16);     // dim1 hi=0 | tile_dim0 = 32 (127:112)
            g1[4] = 128;             // tile_dim1 = 128, tile_dim2 = 0
            g1[5] = K;               // tensor_dim0_stride[31:0] = K elements
            g1[6] = 0;               // stride hi, dim1_stride lo
            g1[7] = 0;
            i32x4 g2 = {0, 0, 0, 0}, g3 = {0, 0, 0, 0};
            i32x8 g4 = {0, 0, 0, 0, 0, 0, 0, 0};
            __builtin_amdgcn_tensor_load_to_lds(g0, g1, g2, g3, g4, 0);
        }
#else
        {
            const __bf16* ap = A + (size_t)(m0 + r2) * K + k0 + c2;
            *(v8bf*)&As[r2][c2]     = *(const v8bf*)ap;
            *(v8bf*)&As[r2][c2 + 8] = *(const v8bf*)(ap + 8);
        }
#endif
        // W tile fp32 -> bf16 (hw cvt), transposed store [n][k]
        {
            const float* wp = W + (size_t)(k0 + kk) * N + n0 + nn;
            #pragma unroll
            for (int j = 0; j < 8; ++j) Bs[nn + j][kk] = f2bf(wp[j]);
        }
#if USE_TDM
        if (wid == 0) __builtin_amdgcn_s_wait_tensorcnt(0);
#endif
        __syncthreads();

        // fragments: A lane(m=ml,+16) k = e + 8*(e>=8) + 8*hlf  -> two 16B loads
        //            B lane(n=ml,+16) k = e + 8*(e>=8) + 8*hlf  -> two 16B loads
        const int ra0 = wy * 32 + ml, ra1 = ra0 + 16;
        const int cb0 = wx * 32 + ml, cb1 = cb0 + 16;
        v16bf a0 = frag16(&As[ra0][8 * hlf], &As[ra0][16 + 8 * hlf]);
        v16bf a1 = frag16(&As[ra1][8 * hlf], &As[ra1][16 + 8 * hlf]);
        v16bf b0 = frag16(&Bs[cb0][8 * hlf], &Bs[cb0][16 + 8 * hlf]);
        v16bf b1 = frag16(&Bs[cb1][8 * hlf], &Bs[cb1][16 + 8 * hlf]);

        acc00 = __builtin_amdgcn_wmma_f32_16x16x32_bf16(false, a0, false, b0, (short)0, acc00, false, false);
        acc01 = __builtin_amdgcn_wmma_f32_16x16x32_bf16(false, a0, false, b1, (short)0, acc01, false, false);
        acc10 = __builtin_amdgcn_wmma_f32_16x16x32_bf16(false, a1, false, b0, (short)0, acc10, false, false);
        acc11 = __builtin_amdgcn_wmma_f32_16x16x32_bf16(false, a1, false, b1, (short)0, acc11, false, false);
        __syncthreads();
    }

    // epilogue: row = m0 + wy*32 + my*16 + r + 8*hlf, col = n0 + wx*32 + nx*16 + ml
    #pragma unroll
    for (int my = 0; my < 2; ++my) {
        #pragma unroll
        for (int nx = 0; nx < 2; ++nx) {
            v8f acc = (my == 0) ? (nx == 0 ? acc00 : acc01)
                                : (nx == 0 ? acc10 : acc11);
            int col = n0 + wx * 32 + nx * 16 + ml;
            float bv = (flags & FLAG_BIAS) ? bias[col] : 0.0f;
            #pragma unroll
            for (int r = 0; r < 8; ++r) {
                int row = m0 + wy * 32 + my * 16 + r + 8 * hlf;
                float v = acc[r] + bv;
                if (flags & FLAG_GELU) v = gelu_exact(v);
                size_t idx = (size_t)row * N + col;
                if (flags & FLAG_OUTBF16) {
                    Cb[idx] = f2bf(v);
                } else {
                    if (flags & FLAG_RES) v += resid[idx];
                    C[idx] = v;
                }
            }
        }
    }
}

// ---------------------------------------------------------------
// timestep embedding + SiLU (per batch; t constant along seq)
__global__ void temb_silu_kernel(const int* __restrict__ t, float* __restrict__ st)
{
    int i = blockIdx.x * blockDim.x + threadIdx.x;
    if (i >= BATCH * DIM) return;
    int b = i / DIM, d = i - b * DIM;
    float tv = (float)t[b * SEQ];
    int j = (d < 384) ? d : d - 384;
    float freq = __expf(-9.210340371976184f * (float)j / 383.0f);
    float ang = tv * freq;
    float v = (d < 384) ? sinf(ang) : cosf(ang);
    st[i] = v / (1.0f + __expf(-v));   // silu
}

// e[b][n] = bias[n] + sum_k st[b][k] * w[k][n]   (M=4 modulation GEMM)
__global__ void adaln_e_kernel(const float* __restrict__ st, const float* __restrict__ w,
                               const float* __restrict__ bias, float* __restrict__ e, int N2)
{
    int i = blockIdx.x * blockDim.x + threadIdx.x;
    if (i >= BATCH * N2) return;
    int b = i / N2, n = i - b * N2;
    const float* sr = st + b * DIM;
    float acc = bias[n];
    for (int k = 0; k < DIM; ++k) acc += sr[k] * w[(size_t)k * N2 + n];
    e[i] = acc;
}

// fused LayerNorm + adaLN modulation; bf16 output when outb != nullptr
__global__ __launch_bounds__(256) void ln_mod_kernel(
    const float* __restrict__ x, const float* __restrict__ g, const float* __restrict__ beta,
    const float* __restrict__ e, float* __restrict__ outf, __bf16* __restrict__ outb)
{
    __shared__ float rs0[256], rs1[256];
    int tok = blockIdx.x, tid = threadIdx.x;
    int batch = tok >> 10;
    const float* xr = x + (size_t)tok * DIM;
    float lx[3], s = 0.f, sq = 0.f;
    #pragma unroll
    for (int j = 0; j < 3; ++j) { float v = xr[tid + j * 256]; lx[j] = v; s += v; sq += v * v; }
    rs0[tid] = s; rs1[tid] = sq; __syncthreads();
    for (int off = 128; off; off >>= 1) {
        if (tid < off) { rs0[tid] += rs0[tid + off]; rs1[tid] += rs1[tid + off]; }
        __syncthreads();
    }
    float mu  = rs0[0] * (1.0f / DIM);
    float var = rs1[0] * (1.0f / DIM) - mu * mu;
    float rstd = rsqrtf(var + 1e-5f);
    const float* sc = e + (size_t)batch * 2 * DIM;
    #pragma unroll
    for (int j = 0; j < 3; ++j) {
        int d = tid + j * 256;
        float y = (lx[j] - mu) * rstd * g[d] + beta[d];
        y = y * (1.0f + sc[d]) + sc[DIM + d];
        size_t idx = (size_t)tok * DIM + d;
        if (outb) outb[idx] = f2bf(y); else outf[idx] = y;
    }
}

// x += h_emb[hpos] + w_emb[wpos]
__global__ __launch_bounds__(256) void posemb_kernel(
    float* __restrict__ x, const int* __restrict__ tids,
    const float* __restrict__ h_emb, const float* __restrict__ w_emb)
{
    int tok = blockIdx.x, tid = threadIdx.x;
    int hp = tids[tok * 3 + 1], wp = tids[tok * 3 + 2];
    #pragma unroll
    for (int j = 0; j < 3; ++j) {
        int d = tid + j * 256;
        x[(size_t)tok * DIM + d] += h_emb[(size_t)hp * DIM + d] + w_emb[(size_t)wp * DIM + d];
    }
}

// fp32 -> bf16 bulk convert
__global__ void cvt_bf16_kernel(const float* __restrict__ in, __bf16* __restrict__ out, int n)
{
    int i = blockIdx.x * blockDim.x + threadIdx.x;
    if (i < n) out[i] = f2bf(in[i]);
}

// ---------------------------------------------------------------
// Local attention: one wave32 per (token, head); lane covers dims
// {lane, lane+32}; <=25 neighbors (5x5 Chebyshev window) inside
// the token's 256-token sample block; online softmax; bf16 out.
// ---------------------------------------------------------------
__global__ __launch_bounds__(256) void local_attn_kernel(
    const float* __restrict__ qkv, const int* __restrict__ tids, __bf16* __restrict__ out)
{
    int g    = blockIdx.x * 8 + (threadIdx.x >> 5);
    int lane = threadIdx.x & 31;
    if (g >= NTOK * NHEAD) return;
    int tok  = g / NHEAD;
    int head = g - tok * NHEAD;
    int sid  = tids[tok * 3];
    __bf16* orow = out + (size_t)tok * DIM + head * DHEAD;
    if (sid == MASKID) { orow[lane] = f2bf(0.f); orow[lane + 32] = f2bf(0.f); return; }
    int hp = tids[tok * 3 + 1], wp = tids[tok * 3 + 2];
    int blk = (tok / 256) * 256;
    const float* qrow = qkv + (size_t)tok * (3 * DIM) + head * DHEAD;
    float q0 = qrow[lane], q1 = qrow[lane + 32];
    float m = -1e30f, ssum = 0.f, a0 = 0.f, a1 = 0.f;
    for (int dh = -2; dh <= 2; ++dh) {
        int nh = hp + dh; if (nh < 0 || nh > 15) continue;
        for (int dw = -2; dw <= 2; ++dw) {
            int nw = wp + dw; if (nw < 0 || nw > 15) continue;
            int ntok = blk + nh * 16 + nw;
            if (tids[ntok * 3] != sid) continue;   // same-sample & not-pad
            const float* krow = qkv + (size_t)ntok * (3 * DIM) + DIM + head * DHEAD;
            float s = q0 * krow[lane] + q1 * krow[lane + 32];
            #pragma unroll
            for (int off = 16; off; off >>= 1) s += __shfl_xor(s, off, 32);
            s *= 0.125f;                           // 1/sqrt(64)
            float nm = fmaxf(m, s);
            float f  = __expf(m - nm), p = __expf(s - nm);
            const float* vrow = krow + DIM;
            a0 = a0 * f + p * vrow[lane];
            a1 = a1 * f + p * vrow[lane + 32];
            ssum = ssum * f + p;
            m = nm;
        }
    }
    float inv = 1.0f / ssum;
    orow[lane] = f2bf(a0 * inv); orow[lane + 32] = f2bf(a1 * inv);
}

// ---------------------------------------------------------------
extern "C" void kernel_launch(void* const* d_in, const int* in_sizes, int n_in,
                              void* d_out, int out_size, void* d_ws, size_t ws_size,
                              hipStream_t stream)
{
    const float* x_in      = (const float*)d_in[0];
    const int*   t_in      = (const int*)  d_in[1];
    const int*   tids      = (const int*)  d_in[2];
    const float* proj_in_w = (const float*)d_in[3];
    const float* proj_in_b = (const float*)d_in[4];
    const float* h_emb     = (const float*)d_in[5];
    const float* w_emb     = (const float*)d_in[6];
    const float* adaln1_w  = (const float*)d_in[7];
    const float* adaln1_b  = (const float*)d_in[8];
    const float* ln1_g     = (const float*)d_in[9];
    const float* ln1_b     = (const float*)d_in[10];
    const float* qkv_w     = (const float*)d_in[11];
    const float* qkv_b     = (const float*)d_in[12];
    const float* o_w       = (const float*)d_in[13];
    const float* o_b       = (const float*)d_in[14];
    const float* adaln2_w  = (const float*)d_in[15];
    const float* adaln2_b  = (const float*)d_in[16];
    const float* ln2_g     = (const float*)d_in[17];
    const float* ln2_b     = (const float*)d_in[18];
    const float* fc1_w     = (const float*)d_in[19];
    const float* fc1_b     = (const float*)d_in[20];
    const float* fc2_w     = (const float*)d_in[21];
    const float* fc2_b     = (const float*)d_in[22];
    const float* nout_w    = (const float*)d_in[23];
    const float* nout_b    = (const float*)d_in[24];
    const float* nout_g    = (const float*)d_in[25];
    const float* nout_beta = (const float*)d_in[26];

    // ---- workspace carve (~69 MB), 256B aligned chunks ----
    char* wsp = (char*)d_ws;
    size_t off = 0;
    auto carve = [&](size_t bytes) { char* p = wsp + off; off += (bytes + 255) & ~(size_t)255; return p; };
    float*  ST  = (float*) carve(BATCH * DIM * 4);               // silu(temb)
    float*  E1  = (float*) carve(BATCH * 2 * DIM * 4);
    float*  E2  = (float*) carve(BATCH * 2 * DIM * 4);
    float*  X   = (float*) carve((size_t)NTOK * DIM * 4);        // residual stream (fp32)
    __bf16* XB  = (__bf16*)carve((size_t)NTOK * DIM * 2);        // bf16 copy of x_in
    __bf16* HB  = (__bf16*)carve((size_t)NTOK * DIM * 2);        // adaLN output (bf16)
    __bf16* AOB = (__bf16*)carve((size_t)NTOK * DIM * 2);        // attention out (bf16)
    char*   SH  = carve((size_t)NTOK * 3 * DIM * 4);             // shared: qkv fp32 / fc1-out bf16
    float*  QKV = (float*)SH;
    __bf16* F1B = (__bf16*)SH;

    dim3 blk(256);
    dim3 g768 (DIM / 64, NTOK / 128);
    dim3 g2304(3 * DIM / 64, NTOK / 128);
    dim3 g3072(HIDN / 64, NTOK / 128);

    temb_silu_kernel<<<(BATCH * DIM + 255) / 256, blk, 0, stream>>>(t_in, ST);
    cvt_bf16_kernel<<<(NTOK * DIM + 255) / 256, blk, 0, stream>>>(x_in, XB, NTOK * DIM);
    gemm_bf16_wmma_kernel<<<g768, blk, 0, stream>>>(
        XB, proj_in_w, proj_in_b, nullptr, X, nullptr, NTOK, DIM, DIM, FLAG_BIAS);
    posemb_kernel<<<NTOK, blk, 0, stream>>>(X, tids, h_emb, w_emb);

    for (int i = 0; i < LNUM; ++i) {
        adaln_e_kernel<<<(BATCH * 2 * DIM + 255) / 256, blk, 0, stream>>>(
            ST, adaln1_w + (size_t)i * DIM * 2 * DIM, adaln1_b + (size_t)i * 2 * DIM, E1, 2 * DIM);
        ln_mod_kernel<<<NTOK, blk, 0, stream>>>(X, ln1_g + (size_t)i * DIM,
                                                ln1_b + (size_t)i * DIM, E1, nullptr, HB);
        gemm_bf16_wmma_kernel<<<g2304, blk, 0, stream>>>(
            HB, qkv_w + (size_t)i * DIM * 3 * DIM, qkv_b + (size_t)i * 3 * DIM,
            nullptr, QKV, nullptr, NTOK, 3 * DIM, DIM, FLAG_BIAS);
        local_attn_kernel<<<(NTOK * NHEAD) / 8, blk, 0, stream>>>(QKV, tids, AOB);
        gemm_bf16_wmma_kernel<<<g768, blk, 0, stream>>>(
            AOB, o_w + (size_t)i * DIM * DIM, o_b + (size_t)i * DIM,
            X, X, nullptr, NTOK, DIM, DIM, FLAG_BIAS | FLAG_RES);
        adaln_e_kernel<<<(BATCH * 2 * DIM + 255) / 256, blk, 0, stream>>>(
            ST, adaln2_w + (size_t)i * DIM * 2 * DIM, adaln2_b + (size_t)i * 2 * DIM, E2, 2 * DIM);
        ln_mod_kernel<<<NTOK, blk, 0, stream>>>(X, ln2_g + (size_t)i * DIM,
                                                ln2_b + (size_t)i * DIM, E2, nullptr, HB);
        gemm_bf16_wmma_kernel<<<g3072, blk, 0, stream>>>(
            HB, fc1_w + (size_t)i * DIM * HIDN, fc1_b + (size_t)i * HIDN,
            nullptr, nullptr, F1B, NTOK, HIDN, DIM, FLAG_BIAS | FLAG_GELU | FLAG_OUTBF16);
        gemm_bf16_wmma_kernel<<<g768, blk, 0, stream>>>(
            F1B, fc2_w + (size_t)i * HIDN * DIM, fc2_b + (size_t)i * DIM,
            X, X, nullptr, NTOK, DIM, HIDN, FLAG_BIAS | FLAG_RES);
    }

    adaln_e_kernel<<<(BATCH * 2 * DIM + 255) / 256, blk, 0, stream>>>(
        ST, nout_w, nout_b, E1, 2 * DIM);
    ln_mod_kernel<<<NTOK, blk, 0, stream>>>(X, nout_g, nout_beta, E1, (float*)d_out, nullptr);
}
